// WaveRNN_13640816132166
// MI455X (gfx1250) — compile-verified, matrix operand order checked by reading, not run on hardware
//
#include <hip/hip_runtime.h>
#include <math.h>

typedef _Float16 h16;
typedef __attribute__((ext_vector_type(16))) _Float16 v16h;
typedef __attribute__((ext_vector_type(8)))  float    v8f;

#define BB 8
#define TT 512
#define MM 80
#define UU 896
#define CC 256
#define U3 (3*UU)          // 2688
#define KT_R (UU/32)       // 28 k-tiles over hidden
#define NT_R (U3/16)       // 168 n-tiles over 3U
#define KT_X 3             // Wk K padded 83 -> 96
#define KT_HD ((UU/2)/32)  // 14
#define NT_HD1 ((UU/2)/16) // 28
#define NT_HD2 (CC/16)     // 16
#define NWG 8
#define NTHREADS 256
#define NWAVES 8
#define JT_PER_WG (56/NWG) // 7 h-col tiles per WG
#define NUNITS (JT_PER_WG*3)

// ---- WMMA 16-bit tile position: element (row/col rc 0..15, kk 0..31) -> byte-pair slot
static __device__ __forceinline__ int tpos(int rc, int kk) {
  int g, v, p;
  if (kk < 16) { g = kk >> 3; int q = kk & 7; v = q >> 1; p = q & 1; }
  else { int k2 = kk - 16; g = k2 >> 3; int q = k2 & 7; v = 4 + (q >> 1); p = q & 1; }
  return (rc + 16*g)*16 + v*2 + p;
}

static __device__ __forceinline__ float rngu(unsigned t, unsigned pass, unsigned m, unsigned i) {
  unsigned x = t*0x9E3779B9u ^ (pass+1u)*0x85EBCA6Bu ^ (m+1u)*0xC2B2AE35u ^ (i+1u)*0x27D4EB2Fu;
  x ^= x >> 16; x *= 0x7FEB352Du; x ^= x >> 15; x *= 0x846CA68Bu; x ^= x >> 16;
  return ((x >> 8) + 0.5f) * (1.0f/16777216.0f);
}

// sense-reversing global barrier, alternating counters; bar[0],bar[1]=cnt slots, bar[2]=gen
static __device__ void gbar(unsigned* bar, unsigned id) {
  __syncthreads();
  if (threadIdx.x == 0) {
    __threadfence();
    volatile unsigned* gen = bar + 2;
    unsigned* cnt = bar + (id & 1u);
    unsigned g = *gen;
    if (atomicAdd(cnt, 1u) == (unsigned)(NWG - 1)) {
      atomicExch(cnt, 0u);
      __threadfence();
      atomicAdd((unsigned*)gen, 1u);
    } else {
      while (*gen == g) __builtin_amdgcn_s_sleep(1);
      __threadfence();
    }
  }
  __syncthreads();
}

// ---- pack f32 weight matrix [Krows x N] into WMMA f16 B-tiles, tile=(ntile*KT+kt)*512 halves
__global__ void pack_b_tiles(const float* __restrict__ W, h16* __restrict__ dst,
                             int Krows, int KT, int N) {
  int tile = blockIdx.x;
  int kt = tile % KT;
  int ntile = tile / KT;
  h16* out = dst + (size_t)tile * 512;
  for (int e = threadIdx.x; e < 512; e += blockDim.x) {
    int lane = e >> 4, hh = e & 15;
    int n = lane & 15, g = lane >> 4;
    int v = hh >> 1, p = hh & 1;
    int kk = (v < 4) ? (v*2 + g*8 + p) : (16 + (v-4)*2 + g*8 + p);
    int k = kt*32 + kk;
    int col = ntile*16 + n;
    float val = (k < Krows) ? W[(size_t)k*N + col] : 0.0f;
    out[e] = (h16)val;
  }
}

__global__ void zero_u32(unsigned* __restrict__ p, size_t n) {
  for (size_t i = blockIdx.x*(size_t)blockDim.x + threadIdx.x; i < n;
       i += (size_t)gridDim.x*blockDim.x) p[i] = 0u;
}

// head: D = relu(A@W1+b1) @ W2 + b2, A = 16x448 f16 tiles (global), logits rows 0..7 to LDS
static __device__ void head_gemm(const h16* __restrict__ inA,
                                 const h16* __restrict__ W1T, const float* __restrict__ b1,
                                 const h16* __restrict__ W2T, const float* __restrict__ b2,
                                 h16* s_t1, float* s_logit, int wave, int lane) {
  for (int nt = wave; nt < NT_HD1; nt += NWAVES) {
    v8f acc = {};
    const h16* wb = W1T + (size_t)nt*KT_HD*512;
    for (int kt = 0; kt < KT_HD; ++kt) {
      v16h a = *(const v16h*)(inA + (size_t)kt*512 + lane*16);
      v16h b = *(const v16h*)(wb + (size_t)kt*512 + lane*16);
      acc = __builtin_amdgcn_wmma_f32_16x16x32_f16(false, a, false, b, (short)0, acc, false, false);
    }
    int n = lane & 15, g = lane >> 4;
    float bias = b1[nt*16 + n];
    for (int r = 0; r < 8; ++r) {
      int mrow = r + 8*g;
      float vr = acc[r] + bias; vr = vr > 0.f ? vr : 0.f;
      int col = nt*16 + n;
      s_t1[(col >> 5)*512 + tpos(mrow, col & 31)] = (h16)vr;
    }
  }
  __syncthreads();
  for (int nt = wave; nt < NT_HD2; nt += NWAVES) {
    v8f acc = {};
    const h16* wb = W2T + (size_t)nt*KT_HD*512;
    for (int kt = 0; kt < KT_HD; ++kt) {
      v16h a = *(const v16h*)(s_t1 + kt*512 + lane*16);
      v16h b = *(const v16h*)(wb + (size_t)kt*512 + lane*16);
      acc = __builtin_amdgcn_wmma_f32_16x16x32_f16(false, a, false, b, (short)0, acc, false, false);
    }
    if (lane < 16) {
      float bias = b2[nt*16 + lane];
      for (int r = 0; r < 8; ++r) s_logit[r*CC + nt*16 + lane] = acc[r] + bias;
    }
  }
  __syncthreads();
}

// softmax over 256 logits (row), then Gumbel-argmax over the *probabilities* (reference
// feeds softmax probs into categorical). Deterministic hash RNG keyed by (t,pass,m,idx).
static __device__ int sample_row(const float* logits, int t, int pass, int m, int lane) {
  float v[8];
  float mx = -1e30f;
  for (int i = 0; i < 8; ++i) { v[i] = logits[lane + 32*i]; mx = fmaxf(mx, v[i]); }
  for (int o = 16; o > 0; o >>= 1) mx = fmaxf(mx, __shfl_xor(mx, o, 32));
  float s = 0.f;
  for (int i = 0; i < 8; ++i) { v[i] = expf(v[i] - mx); s += v[i]; }
  for (int o = 16; o > 0; o >>= 1) s += __shfl_xor(s, o, 32);
  float inv = 1.f / s;
  float best = -1e30f; int bidx = 0;
  for (int i = 0; i < 8; ++i) {
    int idx = lane + 32*i;
    float u = rngu((unsigned)t, (unsigned)pass, (unsigned)m, (unsigned)idx);
    float score = v[i]*inv + (-logf(-logf(u)));
    if (score > best) { best = score; bidx = idx; }
  }
  for (int o = 16; o > 0; o >>= 1) {
    float ob = __shfl_xor(best, o, 32);
    int   oi = __shfl_xor(bidx, o, 32);
    if (ob > best || (ob == best && oi < bidx)) { best = ob; bidx = oi; }
  }
  return bidx;
}

__global__ __launch_bounds__(NTHREADS, 1)
void wavernn_main(const float* __restrict__ mels, const float* __restrict__ Wk,
                  const float* __restrict__ bg,
                  const float* __restrict__ b1c, const float* __restrict__ b2c,
                  const float* __restrict__ b1f, const float* __restrict__ b2f,
                  const h16* __restrict__ WrT, const h16* __restrict__ WkT,
                  const h16* __restrict__ W1cT, const h16* __restrict__ W2cT,
                  const h16* __restrict__ W1fT, const h16* __restrict__ W2fT,
                  h16* __restrict__ hA, h16* __restrict__ h0A, h16* __restrict__ h1A,
                  float* __restrict__ hPlain, float* cv, float* fv,
                  unsigned* bar, float* __restrict__ out) {
  __shared__ h16 s_mi[NUNITS][8][16];
  __shared__ h16 s_mx[NUNITS][8][16];
  __shared__ __align__(32) h16 s_xA[KT_X*512];
  __shared__ __align__(32) h16 s_t1[KT_HD*512];
  __shared__ float s_logit[8*CC];

  const int tid  = threadIdx.x;
  const int wave = tid >> 5;
  const int lane = tid & 31;
  const int wg   = blockIdx.x;
  unsigned barid = 0;
  const float* wkrow = Wk + (size_t)(MM + 2) * U3;   // row 82 (coarse-feedback channel)

  for (int t = 0; t < TT; ++t) {
    int newc = 0;
    // ---- build x0 A-tiles (mel | coarse | fine | 0), rows 8..15 zero pad
    for (int e = tid; e < KT_X*512; e += NTHREADS) {
      int kt = e >> 9, r = e & 511;
      int ln = r >> 4, hh = r & 15;
      int m = ln & 15, g = ln >> 4;
      int v = hh >> 1, p = hh & 1;
      int kk = (v < 4) ? (v*2 + g*8 + p) : (16 + (v-4)*2 + g*8 + p);
      int k = kt*32 + kk;
      float val = 0.f;
      if (m < BB) {
        if (k < MM) val = mels[(size_t)m*TT*MM + (size_t)t*MM + k];
        else if (k == MM)     val = ((volatile float*)cv)[m];
        else if (k == MM + 1) val = ((volatile float*)fv)[m];
      }
      s_xA[e] = (h16)val;
    }
    __syncthreads();

    // ---- phase 1: this WG's slice of mi = h@Wr + b[1] and mx0 = x0@Wk + b[0]
    for (int u = wave; u < NUNITS; u += NWAVES) {
      int jl = u / 3, g = u % 3;
      int j = wg*JT_PER_WG + jl;
      int ntile = g*56 + j;
      int ncol0 = ntile*16;
      v8f acc = {};
      const h16* wb = WrT + (size_t)ntile*KT_R*512;
      for (int kt = 0; kt < KT_R; ++kt) {
        v16h a = *(const v16h*)(hA + (size_t)kt*512 + lane*16);
        v16h b = *(const v16h*)(wb + (size_t)kt*512 + lane*16);
        acc = __builtin_amdgcn_wmma_f32_16x16x32_f16(false, a, false, b, (short)0, acc, false, false);
      }
      if (lane < 16) {
        float bias = bg[U3 + ncol0 + lane];
        for (int r = 0; r < 8; ++r) s_mi[u][r][lane] = (h16)(acc[r] + bias);
      }
      v8f acx = {};
      const h16* wkb = WkT + (size_t)ntile*KT_X*512;
      for (int kt = 0; kt < KT_X; ++kt) {
        v16h a = *(const v16h*)(s_xA + kt*512 + lane*16);
        v16h b = *(const v16h*)(wkb + (size_t)kt*512 + lane*16);
        acx = __builtin_amdgcn_wmma_f32_16x16x32_f16(false, a, false, b, (short)0, acx, false, false);
      }
      if (lane < 16) {
        float bias = bg[ncol0 + lane];
        for (int r = 0; r < 8; ++r) s_mx[u][r][lane] = (h16)(acx[r] + bias);
      }
    }
    __syncthreads();

    // ---- phase 1 gates -> h0 (only cols < U/2 feed the coarse head)
    for (int e = tid; e < JT_PER_WG*16*8; e += NTHREADS) {
      int jl = e >> 7;
      int n  = (e >> 3) & 15;
      int m  = e & 7;
      int col = (wg*JT_PER_WG + jl)*16 + n;
      float miz = (float)s_mi[jl*3+0][m][n], mir = (float)s_mi[jl*3+1][m][n], mih = (float)s_mi[jl*3+2][m][n];
      float mxz = (float)s_mx[jl*3+0][m][n], mxr = (float)s_mx[jl*3+1][m][n], mxh = (float)s_mx[jl*3+2][m][n];
      float z  = 1.f/(1.f + expf(-(mxz + miz)));
      float rr = 1.f/(1.f + expf(-(mxr + mir)));
      float hh = tanhf(mxh + rr*mih);
      float hc = hPlain[m*UU + col];
      float h0 = z*hc + (1.f - z)*hh;
      if (col < UU/2) h0A[(col >> 5)*512 + tpos(m, col & 31)] = (h16)h0;
    }
    __threadfence();
    gbar(bar, barid++);

    // ---- coarse head + sample (WG0 only)
    if (wg == 0) {
      head_gemm(h0A, W1cT, b1c, W2cT, b2c, s_t1, s_logit, wave, lane);
      if (wave < BB) {
        int idx = sample_row(s_logit + wave*CC, t, 0, wave, lane);
        if (lane == 0) {
          newc = idx;
          cv[wave] = (float)idx * (2.f/255.f) - 1.f;   // new coarse val, read by all WGs
        }
      }
    }
    __threadfence();
    gbar(bar, barid++);

    // ---- phase 2: rank-1 fine-pass correction, full GRU update -> h1 (next hidden)
    for (int e = tid; e < JT_PER_WG*16*8; e += NTHREADS) {
      int jl = e >> 7;
      int n  = (e >> 3) & 15;
      int m  = e & 7;
      int col = (wg*JT_PER_WG + jl)*16 + n;
      float nc = ((volatile float*)cv)[m];
      float miz = (float)s_mi[jl*3+0][m][n], mir = (float)s_mi[jl*3+1][m][n], mih = (float)s_mi[jl*3+2][m][n];
      float mxz = (float)s_mx[jl*3+0][m][n] + nc * wkrow[col];
      float mxr = (float)s_mx[jl*3+1][m][n] + nc * wkrow[UU + col];
      float mxh = (float)s_mx[jl*3+2][m][n] + nc * wkrow[2*UU + col];
      float z  = 1.f/(1.f + expf(-(mxz + miz)));
      float rr = 1.f/(1.f + expf(-(mxr + mir)));
      float hh = tanhf(mxh + rr*mih);
      float hc = hPlain[m*UU + col];
      float h1 = z*hc + (1.f - z)*hh;
      hPlain[m*UU + col] = h1;
      hA[(col >> 5)*512 + tpos(m, col & 31)] = (h16)h1;
      if (col >= UU/2) {
        int c2 = col - UU/2;
        h1A[(c2 >> 5)*512 + tpos(m, c2 & 31)] = (h16)h1;
      }
    }
    __threadfence();
    gbar(bar, barid++);

    // ---- fine head + sample + emit (WG0 only)
    if (wg == 0) {
      head_gemm(h1A, W1fT, b1f, W2fT, b2f, s_t1, s_logit, wave, lane);
      if (wave < BB) {
        int idx = sample_row(s_logit + wave*CC, t, 1, wave, lane);
        if (lane == 0) {
          fv[wave] = (float)idx * (2.f/255.f) - 1.f;
          out[(size_t)wave*TT + t] = (float)(newc*CC + idx) / 32767.5f - 1.f;
        }
      }
    }
    __threadfence();
    gbar(bar, barid++);
  }
}

extern "C" void kernel_launch(void* const* d_in, const int* in_sizes, int n_in,
                              void* d_out, int out_size, void* d_ws, size_t ws_size,
                              hipStream_t stream) {
  (void)in_sizes; (void)n_in; (void)out_size; (void)ws_size;
  const float* mels = (const float*)d_in[0];
  const float* Wk   = (const float*)d_in[1];
  const float* Wr   = (const float*)d_in[2];
  const float* bg   = (const float*)d_in[3];
  const float* W1c  = (const float*)d_in[4];
  const float* b1c  = (const float*)d_in[5];
  const float* W2c  = (const float*)d_in[6];
  const float* b2c  = (const float*)d_in[7];
  const float* W1f  = (const float*)d_in[8];
  const float* b1f  = (const float*)d_in[9];
  const float* W2f  = (const float*)d_in[10];
  const float* b2f  = (const float*)d_in[11];
  float* out = (float*)d_out;

  // workspace layout (half regions are 1KB-tile aligned)
  constexpr size_t WrT_H  = (size_t)NT_R  * KT_R  * 512; // 2,408,448
  constexpr size_t WkT_H  = (size_t)NT_R  * KT_X  * 512; //   258,048
  constexpr size_t W1T_H  = (size_t)NT_HD1* KT_HD * 512; //   200,704
  constexpr size_t W2T_H  = (size_t)NT_HD2* KT_HD * 512; //   114,688
  constexpr size_t hA_H   = (size_t)KT_R  * 512;         //    14,336
  constexpr size_t hHd_H  = (size_t)KT_HD * 512;         //     7,168

  h16* WrT  = (h16*)d_ws;
  h16* WkT  = WrT  + WrT_H;
  h16* W1cT = WkT  + WkT_H;
  h16* W2cT = W1cT + W1T_H;
  h16* W1fT = W2cT + W2T_H;
  h16* W2fT = W1fT + W1T_H;
  h16* hA   = W2fT + W2T_H;
  h16* h0A  = hA   + hA_H;
  h16* h1A  = h0A  + hHd_H;
  float* hPlain = (float*)(h1A + hHd_H);
  float* cv = hPlain + BB*UU;
  float* fv = cv + BB;
  unsigned* bar = (unsigned*)(fv + BB);

  // zero dynamic state: hA/h0A/h1A (f16), hPlain, cv, fv, bar
  size_t zero_u32_count = ((hA_H + 2*hHd_H) * sizeof(h16) + BB*UU*sizeof(float)
                           + 2*BB*sizeof(float) + 16*sizeof(unsigned)) / 4;
  zero_u32<<<64, 256, 0, stream>>>((unsigned*)hA, zero_u32_count);

  // pack weights into WMMA f16 B-tile layout
  pack_b_tiles<<<NT_R  * KT_R , 256, 0, stream>>>(Wr,  WrT,  UU,    KT_R,  U3);
  pack_b_tiles<<<NT_R  * KT_X , 256, 0, stream>>>(Wk,  WkT,  MM+3,  KT_X,  U3);
  pack_b_tiles<<<NT_HD1* KT_HD, 256, 0, stream>>>(W1c, W1cT, UU/2,  KT_HD, UU/2);
  pack_b_tiles<<<NT_HD2* KT_HD, 256, 0, stream>>>(W2c, W2cT, UU/2,  KT_HD, CC);
  pack_b_tiles<<<NT_HD1* KT_HD, 256, 0, stream>>>(W1f, W1fT, UU/2,  KT_HD, UU/2);
  pack_b_tiles<<<NT_HD2* KT_HD, 256, 0, stream>>>(W2f, W2fT, UU/2,  KT_HD, CC);

  // persistent cooperative sequential scan
  wavernn_main<<<NWG, NTHREADS, 0, stream>>>(mels, Wk, bg, b1c, b2c, b1f, b2f,
                                             WrT, WkT, W1cT, W2cT, W1fT, W2fT,
                                             hA, h0A, h1A, hPlain, cv, fv, bar, out);
}
